// GCN_Graph_59313498358439
// MI455X (gfx1250) — matrix operand order repaired and checked by reading, not compiled, not run on hardware
//
#include <hip/hip_runtime.h>
#include <hip/hip_bf16.h>

typedef __attribute__((ext_vector_type(16))) _Float16 v16h;
typedef __attribute__((ext_vector_type(8)))  float    v8f;

#define HD    128
#define PADW  136        // padded LDS row stride (halfs): 272B -> conflict-free lanes
#define NFEA  9
#define VOC   119
#define BNEPS 1e-5f

// ---------------------------------------------------------------- utilities
__global__ void fill0_kernel(float* __restrict__ p, long n) {
    long i = (long)blockIdx.x * blockDim.x + threadIdx.x;
    if (i < n) p[i] = 0.0f;
}

__device__ __forceinline__ float atomAddF(float* p, float v) {
    return unsafeAtomicAdd(p, v);   // native global_atomic_add_f32 on gfx1250
}

// ---------------------------------------------------------------- degree / dinv
__global__ void deg_kernel(const int* __restrict__ dst, float* __restrict__ deg, int E) {
    int e = blockIdx.x * blockDim.x + threadIdx.x;
    if (e < E) atomAddF(&deg[dst[e]], 1.0f);
}

__global__ void dinv_kernel(float* __restrict__ deg, int N) {
    int i = blockIdx.x * blockDim.x + threadIdx.x;
    if (i < N) deg[i] = __frsqrt_rn(1.0f + deg[i]);   // includes self-loop
}

// ---------------------------------------------------------------- atom encoder
// h[n,:] = sum_f emb[f, x[n,f], :]   (float4 vectorized: 32 quads per node)
__global__ void encoder_kernel(const int* __restrict__ x, const float* __restrict__ emb,
                               float* __restrict__ h, int N) {
    long i = (long)blockIdx.x * blockDim.x + threadIdx.x;   // over N*32 quads
    if (i >= (long)N * (HD / 4)) return;
    int n = (int)(i >> 5);
    int q = (int)(i & 31);
    const float4* e4 = (const float4*)emb;
    float4 s = make_float4(0.f, 0.f, 0.f, 0.f);
    #pragma unroll
    for (int f = 0; f < NFEA; ++f) {
        int idx = x[n * NFEA + f];
        float4 t = e4[(long)(f * VOC + idx) * (HD / 4) + q];
        s.x += t.x; s.y += t.y; s.z += t.z; s.w += t.w;
    }
    ((float4*)h)[i] = s;
}

// ---------------------------------------------------------------- WMMA GEMM
// out[N,128] = h[N,128] @ W[128,128]   (f16 multiply, f32 accumulate)
// LDS: W transposed (WT[n][k]) so B-fragments are contiguous b128 loads;
//      both tiles padded to PADW halfs/row to kill bank conflicts.
__global__ __launch_bounds__(256) void gemm_kernel(const float* __restrict__ hin,
                                                   const float* __restrict__ W,
                                                   float* __restrict__ out, int nrows) {
    __shared__ _Float16 WT[HD * PADW];   // ~34 KB, transposed weights
    __shared__ _Float16 hl[64 * PADW];   // ~17 KB, h tile
    const int tid = threadIdx.x;
    // stage W transposed (fp32 -> f16): WT[n*PADW + k] = W[k*HD + n]
    for (int i = tid; i < HD * HD; i += 256) {
        int k = i >> 7, n = i & (HD - 1);
        WT[n * PADW + k] = (_Float16)W[i];
    }
    // stage 64-row h tile (fp32 -> f16), zero-pad tail rows
    const int rowbase = blockIdx.x * 64;
    for (int i = tid; i < 64 * HD; i += 256) {
        int r = i >> 7, c = i & (HD - 1);
        int gr = rowbase + r;
        hl[r * PADW + c] = (_Float16)((gr < nrows) ? hin[(long)gr * HD + c] : 0.0f);
    }
    __syncthreads();

    const int wave = tid >> 5;           // 8 waves, each owns 16 output columns
    const int lane = tid & 31;
    const int colbase = wave * 16;
    const int l  = lane & 15;
    const bool hi = lane >= 16;

    // B fragments for all 4 K-steps: element e <-> K = kb+e (contiguous in WT)
    v16h bfrag[4];
    #pragma unroll
    for (int kt = 0; kt < 4; ++kt) {
        const int kb = kt * 32 + (hi ? 16 : 0);
        const _Float16* bp = &WT[(colbase + l) * PADW + kb];
        #pragma unroll
        for (int e = 0; e < 16; ++e) bfrag[kt][e] = bp[e];
    }

    #pragma unroll
    for (int mt = 0; mt < 4; ++mt) {
        v8f acc = {};
        const _Float16* ap0 = &hl[(mt * 16 + l) * PADW];
        #pragma unroll
        for (int kt = 0; kt < 4; ++kt) {
            const int ka = kt * 32 + (hi ? 16 : 0);
            v16h afrag;
            #pragma unroll
            for (int e = 0; e < 8; ++e) afrag[e]     = ap0[ka + e];
            #pragma unroll
            for (int e = 0; e < 8; ++e) afrag[8 + e] = ap0[ka + 8 + e];
            acc = __builtin_amdgcn_wmma_f32_16x16x32_f16(
                      false, afrag, false, bfrag[kt], (short)0, acc, false, false);
        }
        const int rbase = rowbase + mt * 16 + (hi ? 8 : 0);
        float* op = out + (long)rbase * HD + colbase + l;
        if (rowbase + 64 <= nrows) {        // full tile: branch-free stores
            #pragma unroll
            for (int r = 0; r < 8; ++r) op[(long)r * HD] = acc[r];
        } else {
            #pragma unroll
            for (int r = 0; r < 8; ++r)
                if (rbase + r < nrows) op[(long)r * HD] = acc[r];
        }
    }
}

// ---------------------------------------------------------------- self-loop init
// agg[n,c] = hw[n,c] * dinv[n]^2 + b[c]   (replaces fill0 + later re-read of hw)
__global__ void selfinit_kernel(float* __restrict__ agg, const float* __restrict__ hw,
                                const float* __restrict__ dinv, const float* __restrict__ b,
                                int N) {
    long i = (long)blockIdx.x * blockDim.x + threadIdx.x;
    if (i >= (long)N * HD) return;
    int n = (int)(i >> 7);
    int c = (int)(i & (HD - 1));
    float di = dinv[n];
    agg[i] = hw[i] * di * di + b[c];
}

// ---------------------------------------------------------------- edge scatter
// agg[dst] += hw[src] * dinv[src]*dinv[dst]   (one wave32 per edge, 4 cols/lane)
__global__ void scatter_kernel(const float* __restrict__ hw, const int* __restrict__ src,
                               const int* __restrict__ dst, const float* __restrict__ dinv,
                               float* __restrict__ agg, int E) {
    int e = blockIdx.x * (blockDim.x >> 5) + (threadIdx.x >> 5);
    if (e >= E) return;
    int lane = threadIdx.x & 31;
    int s = src[e], d = dst[e];
    float nrm = dinv[s] * dinv[d];
    #pragma unroll
    for (int i = 0; i < 4; ++i) {
        int c = lane + i * 32;
        atomAddF(&agg[(long)d * HD + c], hw[(long)s * HD + c] * nrm);
    }
}

// ---------------------------------------------------------------- BN + ReLU (in place)
__global__ void bn_relu_kernel(float* __restrict__ h, const float* __restrict__ g,
                               const float* __restrict__ be, const float* __restrict__ m,
                               const float* __restrict__ v, long NH) {
    long i = (long)blockIdx.x * blockDim.x + threadIdx.x;
    if (i >= NH) return;
    int c = (int)(i & (HD - 1));
    float val = h[i];
    val = g[c] * (val - m[c]) * __frsqrt_rn(v[c] + BNEPS) + be[c];
    h[i] = fmaxf(val, 0.0f);
}

// ---------------------------------------------------------------- pooling
__global__ void count_kernel(const int* __restrict__ batch, float* __restrict__ counts, int N) {
    int i = blockIdx.x * blockDim.x + threadIdx.x;
    if (i < N) atomAddF(&counts[batch[i]], 1.0f);
}

__global__ void pool_kernel(const float* __restrict__ h, const int* __restrict__ batch,
                            float* __restrict__ sums, int N) {
    int n = blockIdx.x * (blockDim.x >> 5) + (threadIdx.x >> 5);
    if (n >= N) return;
    int lane = threadIdx.x & 31;
    int b = batch[n];
    #pragma unroll
    for (int i = 0; i < 4; ++i) {
        int c = lane + i * 32;
        atomAddF(&sums[(long)b * HD + c], h[(long)n * HD + c]);
    }
}

// out[g] = (sum[g] . Wh) / max(count,1) + bh
__global__ __launch_bounds__(128) void final_kernel(const float* __restrict__ sums,
                                                    const float* __restrict__ counts,
                                                    const float* __restrict__ Wh,
                                                    const float* __restrict__ bh,
                                                    float* __restrict__ out, int G) {
    __shared__ float red[HD];
    int gph = blockIdx.x, t = threadIdx.x;
    if (gph >= G) return;
    red[t] = sums[(long)gph * HD + t] * Wh[t];
    __syncthreads();
    for (int s = 64; s > 0; s >>= 1) {
        if (t < s) red[t] += red[t + s];
        __syncthreads();
    }
    if (t == 0) out[gph] = red[0] / fmaxf(counts[gph], 1.0f) + bh[0];
}

// ================================================================ launcher
extern "C" void kernel_launch(void* const* d_in, const int* in_sizes, int n_in,
                              void* d_out, int out_size, void* d_ws, size_t ws_size,
                              hipStream_t stream) {
    const int*   x      = (const int*)  d_in[0];
    const int*   eidx   = (const int*)  d_in[1];
    const int*   batch  = (const int*)  d_in[2];
    const float* emb    = (const float*)d_in[3];
    const float* Wv[3]  = { (const float*)d_in[4], (const float*)d_in[6], (const float*)d_in[8] };
    const float* bv[3]  = { (const float*)d_in[5], (const float*)d_in[7], (const float*)d_in[9] };
    const float* gv[2]  = { (const float*)d_in[10], (const float*)d_in[14] };
    const float* bev[2] = { (const float*)d_in[11], (const float*)d_in[15] };
    const float* mv[2]  = { (const float*)d_in[12], (const float*)d_in[16] };
    const float* vv[2]  = { (const float*)d_in[13], (const float*)d_in[17] };
    const float* Whp    = (const float*)d_in[18];
    const float* bhp    = (const float*)d_in[19];
    float* out          = (float*)d_out;

    const int N = in_sizes[2];          // nodes
    const int E = in_sizes[1] / 2;      // edges
    const int G = out_size;             // graphs (OUT == 1)
    const int* src = eidx;
    const int* dst = eidx + E;

    // workspace carve-up
    float* bufP   = (float*)d_ws;                    // N*H  (h / agg, ping)
    float* bufQ   = bufP + (size_t)N * HD;           // N*H  (hw, pong)
    float* dinv   = bufQ + (size_t)N * HD;           // N
    float* counts = dinv + N;                        // G
    float* sums   = counts + G;                      // G*H

    const long NH = (long)N * HD;
    auto cdiv = [](long a, long b) { return (int)((a + b - 1) / b); };

    // degrees -> dinv
    fill0_kernel<<<cdiv(N, 256), 256, 0, stream>>>(dinv, N);
    fill0_kernel<<<cdiv(G, 256), 256, 0, stream>>>(counts, G);
    fill0_kernel<<<cdiv((long)G * HD, 256), 256, 0, stream>>>(sums, (long)G * HD);
    deg_kernel <<<cdiv(E, 256), 256, 0, stream>>>(dst, dinv, E);
    dinv_kernel<<<cdiv(N, 256), 256, 0, stream>>>(dinv, N);

    // atom encoder -> bufP
    encoder_kernel<<<cdiv((long)N * (HD / 4), 256), 256, 0, stream>>>(x, emb, bufP, N);

    // three GCN layers: gemm -> selfloop-init -> scatter -> (BN+ReLU)
    const int gemm_blocks = cdiv(N, 64);
    const int edge_blocks = cdiv(E, 8);     // 8 wave32 edges per 256-thread block
    for (int layer = 0; layer < 3; ++layer) {
        gemm_kernel    <<<gemm_blocks, 256, 0, stream>>>(bufP, Wv[layer], bufQ, N);
        selfinit_kernel<<<cdiv(NH, 256), 256, 0, stream>>>(bufP, bufQ, dinv, bv[layer], N);
        scatter_kernel <<<edge_blocks, 256, 0, stream>>>(bufQ, src, dst, dinv, bufP, E);
        if (layer < 2)
            bn_relu_kernel<<<cdiv(NH, 256), 256, 0, stream>>>(
                bufP, gv[layer], bev[layer], mv[layer], vv[layer], NH);
    }

    // global mean pool + head
    count_kernel<<<cdiv(N, 256), 256, 0, stream>>>(batch, counts, N);
    pool_kernel <<<cdiv(N, 8), 256, 0, stream>>>(bufP, batch, sums, N);
    final_kernel<<<G, 128, 0, stream>>>(sums, counts, Whp, bhp, out, G);
}
